// TransformerXL_61306363183370
// MI455X (gfx1250) — matrix operand order repaired and checked by reading, not compiled, not run on hardware
//
#include <hip/hip_runtime.h>

typedef __bf16 bf16;
typedef bf16  v16bf __attribute__((ext_vector_type(16)));
typedef bf16  v8bf  __attribute__((ext_vector_type(8)));
typedef float v8f   __attribute__((ext_vector_type(8)));

#define NLAYER  8
#define DMODEL  1024
#define NHEAD   16
#define DHEAD   64
#define DINNER  4096
#define QLEN    512
#define MLEN    512
#define BSZ     4
#define KLEN    1024      // MLEN + QLEN
#define NROWS   2048      // QLEN * BSZ
#define CATROWS 4096      // KLEN * BSZ

// ---------------------------------------------------------------------------
// Elementwise helpers
// ---------------------------------------------------------------------------

__global__ __launch_bounds__(256) void cvt_bf16_kernel(const float* __restrict__ in,
                                                       bf16* __restrict__ out, int n) {
  int i = blockIdx.x * 256 + threadIdx.x;
  if (i < n) out[i] = (bf16)in[i];
}

// sinusoidal relative position embedding R, positions klen-1..0, stored bf16
__global__ __launch_bounds__(256) void posemb_kernel(bf16* __restrict__ r) {
  int idx = blockIdx.x * 256 + threadIdx.x;   // KLEN*DMODEL threads
  int j = idx >> 10;
  int i = idx & (DMODEL - 1);
  float pos = (float)(KLEN - 1 - j);
  int t = (i < DMODEL / 2) ? i : (i - DMODEL / 2);
  float invf = __expf(-(2.0f * (float)t / (float)DMODEL) * 9.210340371976184f); // ln(10000)
  float a = pos * invf;
  r[idx] = (bf16)((i < DMODEL / 2) ? sinf(a) : cosf(a));
}

// cat = concat(mem, h) along sequence, converted to bf16 (layouts are flat-compatible)
__global__ __launch_bounds__(256) void cat_kernel(const float* __restrict__ mem,
                                                  const float* __restrict__ h,
                                                  bf16* __restrict__ cat) {
  int idx = blockIdx.x * 256 + threadIdx.x;   // KLEN*BSZ*DMODEL threads
  const int half = MLEN * BSZ * DMODEL;
  float v = (idx < half) ? mem[idx] : h[idx - half];
  cat[idx] = (bf16)v;
}

// ---------------------------------------------------------------------------
// Generic WMMA bf16 GEMM:  C[z] = A[z] (MxK) * W[z] (NxK)^T , fused epilogues
// ---------------------------------------------------------------------------

struct GemmArgs {
  const bf16* A; const bf16* W;
  int M, N, K, lda, ldw;
  long long sAz, sWz; int wzmod;        // batch strides; W batch index = wzmod? z%wzmod : z
  float* outf; bf16* outb; int ldc; long long sCz;
  const float* bias; const float* resid;
  bf16 *qw, *qr, *kk, *vT;              // QKV scatter targets
  const float *rwb, *rrb;               // r_w_bias / r_r_bias (flattened [nh*dh])
};

enum { EPI_F32 = 0, EPI_QKV, EPI_RK, EPI_AV, EPI_RESID, EPI_RELU_BF16, EPI_RESID_BIAS };

template <int EPI>
static __device__ inline void epilog(const GemmArgs& g, int z, int row, int col, float v) {
  if (EPI == EPI_F32) {
    g.outf[(long long)z * g.sCz + (long long)row * g.ldc + col] = v;
  } else if (EPI == EPI_QKV) {
    int j = row / BSZ, b = row - j * BSZ;            // cat row = j*BSZ + b
    int part = col >> 10;                            // 0=q 1=k 2=v  (nh*dh == 1024)
    int hd = col & 1023;
    int n = hd >> 6, dd = hd & 63;
    int zi = b * NHEAD + n;
    if (part == 0) {
      if (j >= MLEN) {
        long long o = ((long long)zi * QLEN + (j - MLEN)) * DHEAD + dd;
        g.qw[o] = (bf16)(v + g.rwb[hd]);             // q + r_w_bias
        g.qr[o] = (bf16)(v + g.rrb[hd]);             // q + r_r_bias
      }
    } else if (part == 1) {
      g.kk[((long long)zi * KLEN + j) * DHEAD + dd] = (bf16)v;
    } else {
      g.vT[((long long)zi * DHEAD + dd) * KLEN + j] = (bf16)v;   // V transposed
    }
  } else if (EPI == EPI_RK) {
    int n = col >> 6, dd = col & 63;
    g.outb[((long long)n * KLEN + row) * DHEAD + dd] = (bf16)v;  // rk[n][klen][dh]
  } else if (EPI == EPI_AV) {
    int b = z / NHEAD, n = z - b * NHEAD;
    g.outb[((long long)row * BSZ + b) * (NHEAD * DHEAD) + n * DHEAD + col] = (bf16)v;
  } else if (EPI == EPI_RESID) {
    long long o = (long long)row * g.ldc + col;
    g.outf[o] = g.resid[o] + v;
  } else if (EPI == EPI_RELU_BF16) {
    float t = v + g.bias[col];
    g.outb[(long long)row * g.ldc + col] = (bf16)(t > 0.f ? t : 0.f);
  } else { // EPI_RESID_BIAS
    long long o = (long long)row * g.ldc + col;
    g.outf[o] = g.resid[o] + v + g.bias[col];
  }
}

static __device__ inline v16bf load_afrag(const bf16* p) {
  v8bf lo = *(const v8bf*)(p);
  v8bf hi = *(const v8bf*)(p + 8);
  return __builtin_shufflevector(lo, hi, 0,1,2,3,4,5,6,7,8,9,10,11,12,13,14,15);
}

// 256 threads = 8 waves, arranged 4 (M) x 2 (N).
// Wave tile: 32 (M) x NSUB*16 (N).  Block tile: 128 (M) x NSUB*32 (N).
// All launches use exact tiling (M % 128 == 0, N % (NSUB*32) == 0) so there are
// no runtime guards: every load and WMMA is unconditional with full EXEC.
template <int EPI, int NSUB>
__global__ __launch_bounds__(256) void gemm_wmma(GemmArgs g) {
  const int lane = threadIdx.x & 31;
  const int wv   = threadIdx.x >> 5;
  const int z    = blockIdx.z;
  const bf16* A = g.A + (long long)z * g.sAz;
  const bf16* W = g.W + (long long)(g.wzmod > 0 ? (z % g.wzmod) : z) * g.sWz;
  const int m0 = blockIdx.y * 128 + (wv & 3) * 32;
  const int n0 = blockIdx.x * (NSUB * 32) + (wv >> 2) * (NSUB * 16);

  const int ksel = lane >> 4;     // lane half selects K sub-range: contiguous k = 16*half + s, same for A and B
  const int l16  = lane & 15;
  const bf16* arow0 = A + (long long)(m0 + l16) * g.lda + (ksel ? 16 : 0);
  const bf16* arow1 = arow0 + (long long)16 * g.lda;
  const bf16* wrow[NSUB];
  #pragma unroll
  for (int t = 0; t < NSUB; ++t)
    wrow[t] = W + (long long)(n0 + t * 16 + l16) * g.ldw + (ksel ? 16 : 0);

  v8f acc[2][NSUB] = {};
  #pragma unroll 2
  for (int k0 = 0; k0 < g.K; k0 += 32) {
    v16bf a0 = load_afrag(arow0 + k0);
    v16bf a1 = load_afrag(arow1 + k0);
    v16bf bb[NSUB];
    #pragma unroll
    for (int t = 0; t < NSUB; ++t) bb[t] = *(const v16bf*)(wrow[t] + k0);
    #pragma unroll
    for (int t = 0; t < NSUB; ++t) {
      acc[0][t] = __builtin_amdgcn_wmma_f32_16x16x32_bf16(false, a0, false, bb[t],
                                                          (short)0, acc[0][t], false, false);
      acc[1][t] = __builtin_amdgcn_wmma_f32_16x16x32_bf16(false, a1, false, bb[t],
                                                          (short)0, acc[1][t], false, false);
    }
  }
  const int rbase = ksel * 8;   // C layout: VGPR i -> M=i (lanes<16) / M=8+i (lanes>=16)
  #pragma unroll
  for (int tm = 0; tm < 2; ++tm) {
    #pragma unroll
    for (int tn = 0; tn < NSUB; ++tn) {
      #pragma unroll
      for (int e = 0; e < 8; ++e) {
        int row = m0 + tm * 16 + rbase + e;
        int col = n0 + tn * 16 + l16;
        epilog<EPI>(g, z, row, col, acc[tm][tn][e]);
      }
    }
  }
}

// ---------------------------------------------------------------------------
// softmax over keys with Transformer-XL rel-shift + causal mask.
// out[i,j] = xp_flat[i*klen + j + qlen]  (derivation of the pad/reshape trick)
// Probabilities written as bf16 aliased over the score row (stride preserved).
// ---------------------------------------------------------------------------
__global__ __launch_bounds__(256) void softmax_kernel(const float* __restrict__ sc,
                                                      const float* __restrict__ bd,
                                                      bf16* __restrict__ p) {
  __shared__ float red[256];
  const int i = blockIdx.x;           // query row
  const int z = blockIdx.y;           // b*NHEAD + n
  const int tid = threadIdx.x;
  const float* scp = sc + ((long long)z * QLEN + i) * KLEN;
  const float* bdz = bd + (long long)z * QLEN * KLEN;
  bf16* pp = p + ((long long)z * QLEN + i) * (2 * KLEN);
  const float scale = 0.125f;         // 1/sqrt(64)
  float s[4];
  #pragma unroll
  for (int u = 0; u < 4; ++u) {
    int j = u * 256 + tid;
    long long F = (long long)i * KLEN + j + QLEN;
    int i2 = (int)(F / (KLEN + 1));
    int jj = (int)(F - (long long)i2 * (KLEN + 1));
    float bdv = (jj == 0) ? 0.0f : bdz[(long long)i2 * KLEN + (jj - 1)];
    float sv = (scp[j] + bdv) * scale;
    if (j > i + MLEN) sv = -3.0e38f;  // causal mask with memory offset
    s[u] = sv;
  }
  float mx = fmaxf(fmaxf(s[0], s[1]), fmaxf(s[2], s[3]));
  red[tid] = mx; __syncthreads();
  for (int st = 128; st > 0; st >>= 1) {
    if (tid < st) red[tid] = fmaxf(red[tid], red[tid + st]);
    __syncthreads();
  }
  mx = red[0]; __syncthreads();
  float e[4], ls = 0.f;
  #pragma unroll
  for (int u = 0; u < 4; ++u) { e[u] = __expf(s[u] - mx); ls += e[u]; }
  red[tid] = ls; __syncthreads();
  for (int st = 128; st > 0; st >>= 1) {
    if (tid < st) red[tid] += red[tid + st];
    __syncthreads();
  }
  float inv = 1.0f / red[0];
  #pragma unroll
  for (int u = 0; u < 4; ++u) pp[u * 256 + tid] = (bf16)(e[u] * inv);
}

// ---------------------------------------------------------------------------
// LayerNorm over d=1024; writes f32 residual stream and bf16 copy for GEMMs
// ---------------------------------------------------------------------------
__global__ __launch_bounds__(256) void ln_kernel(const float* __restrict__ x,
                                                 const float* __restrict__ gg,
                                                 const float* __restrict__ bbias,
                                                 float* __restrict__ h,
                                                 bf16* __restrict__ hb) {
  __shared__ float red[256];
  const int m = blockIdx.x;
  const int tid = threadIdx.x;
  const float* xr = x + (long long)m * DMODEL;
  float v[4]; float s = 0.f;
  #pragma unroll
  for (int u = 0; u < 4; ++u) { v[u] = xr[u * 256 + tid]; s += v[u]; }
  red[tid] = s; __syncthreads();
  for (int st = 128; st > 0; st >>= 1) { if (tid < st) red[tid] += red[tid + st]; __syncthreads(); }
  float mean = red[0] * (1.0f / DMODEL);
  __syncthreads();
  float s2 = 0.f;
  #pragma unroll
  for (int u = 0; u < 4; ++u) { float d = v[u] - mean; s2 += d * d; }
  red[tid] = s2; __syncthreads();
  for (int st = 128; st > 0; st >>= 1) { if (tid < st) red[tid] += red[tid + st]; __syncthreads(); }
  float rstd = rsqrtf(red[0] * (1.0f / DMODEL) + 1e-5f);
  #pragma unroll
  for (int u = 0; u < 4; ++u) {
    int c = u * 256 + tid;
    float y = (v[u] - mean) * rstd * gg[c] + bbias[c];
    h[(long long)m * DMODEL + c] = y;
    hb[(long long)m * DMODEL + c] = (bf16)y;
  }
}

// ---------------------------------------------------------------------------
// Host orchestration
// ---------------------------------------------------------------------------
static inline dim3 gemm_grid(int M, int N, int Z, int nsub) {
  return dim3((unsigned)(N / (nsub * 32)), (unsigned)(M / 128), (unsigned)Z);
}

extern "C" void kernel_launch(void* const* d_in, const int* in_sizes, int n_in,
                              void* d_out, int out_size, void* d_ws, size_t ws_size,
                              hipStream_t stream) {
  (void)in_sizes; (void)n_in; (void)out_size; (void)ws_size;
  const float* w      = (const float*)d_in[0];
  const float* mems   = (const float*)d_in[1];
  const float* rwbias = (const float*)d_in[2];
  const float* rrbias = (const float*)d_in[3];
  const float* qkvw   = (const float*)d_in[4];
  const float* rw     = (const float*)d_in[5];
  const float* ow     = (const float*)d_in[6];
  const float* ln1g   = (const float*)d_in[7];
  const float* ln1b   = (const float*)d_in[8];
  const float* w1     = (const float*)d_in[9];
  const float* b1     = (const float*)d_in[10];
  const float* w2     = (const float*)d_in[11];
  const float* b2     = (const float*)d_in[12];
  const float* ln2g   = (const float*)d_in[13];
  const float* ln2b   = (const float*)d_in[14];

  char* base = (char*)d_ws;
  size_t off = 0;
  auto take = [&](size_t bytes) -> void* {
    void* p = base + off;
    off = (off + bytes + 255) & ~(size_t)255;
    return p;
  };

  bf16* rpe   = (bf16*)take((size_t)KLEN * DMODEL * 2);
  bf16* qkvwb = (bf16*)take((size_t)3 * NHEAD * DHEAD * DMODEL * 2);
  bf16* rwbb  = (bf16*)take((size_t)NHEAD * DHEAD * DMODEL * 2);
  bf16* owb   = (bf16*)take((size_t)DMODEL * NHEAD * DHEAD * 2);
  bf16* w1b   = (bf16*)take((size_t)DINNER * DMODEL * 2);
  bf16* w2b   = (bf16*)take((size_t)DMODEL * DINNER * 2);
  float* h    = (float*)take((size_t)NROWS * DMODEL * 4);
  bf16* hb    = (bf16*)take((size_t)NROWS * DMODEL * 2);
  bf16* catb  = (bf16*)take((size_t)CATROWS * DMODEL * 2);
  bf16* qwb   = (bf16*)take((size_t)BSZ * NHEAD * QLEN * DHEAD * 2);
  bf16* qrb   = (bf16*)take((size_t)BSZ * NHEAD * QLEN * DHEAD * 2);
  bf16* kb    = (bf16*)take((size_t)BSZ * NHEAD * KLEN * DHEAD * 2);
  bf16* vTb   = (bf16*)take((size_t)BSZ * NHEAD * DHEAD * KLEN * 2);
  bf16* rkb   = (bf16*)take((size_t)NHEAD * KLEN * DHEAD * 2);
  float* sc   = (float*)take((size_t)BSZ * NHEAD * QLEN * KLEN * 4);
  float* bd   = (float*)take((size_t)BSZ * NHEAD * QLEN * KLEN * 4);
  bf16* avb   = (bf16*)take((size_t)NROWS * NHEAD * DHEAD * 2);
  float* tmp  = (float*)take((size_t)NROWS * DMODEL * 4);
  bf16* ff1   = (bf16*)take((size_t)NROWS * DINNER * 2);

  posemb_kernel<<<(KLEN * DMODEL) / 256, 256, 0, stream>>>(rpe);
  hipMemcpyAsync(h, w, (size_t)NROWS * DMODEL * 4, hipMemcpyDeviceToDevice, stream);

  for (int l = 0; l < NLAYER; ++l) {
    int n;
    n = 3 * NHEAD * DHEAD * DMODEL;
    cvt_bf16_kernel<<<(n + 255) / 256, 256, 0, stream>>>(qkvw + (size_t)l * n, qkvwb, n);
    n = NHEAD * DHEAD * DMODEL;
    cvt_bf16_kernel<<<(n + 255) / 256, 256, 0, stream>>>(rw + (size_t)l * n, rwbb, n);
    cvt_bf16_kernel<<<(n + 255) / 256, 256, 0, stream>>>(ow + (size_t)l * n, owb, n);
    n = DINNER * DMODEL;
    cvt_bf16_kernel<<<(n + 255) / 256, 256, 0, stream>>>(w1 + (size_t)l * n, w1b, n);
    cvt_bf16_kernel<<<(n + 255) / 256, 256, 0, stream>>>(w2 + (size_t)l * n, w2b, n);

    cat_kernel<<<(CATROWS * DMODEL) / 256, 256, 0, stream>>>(
        mems + (size_t)l * MLEN * BSZ * DMODEL, h, catb);

    { // QKV projection with fused bias-add + head scatter
      GemmArgs g = {};
      g.A = catb; g.W = qkvwb;
      g.M = CATROWS; g.N = 3 * NHEAD * DHEAD; g.K = DMODEL; g.lda = DMODEL; g.ldw = DMODEL;
      g.qw = qwb; g.qr = qrb; g.kk = kb; g.vT = vTb; g.rwb = rwbias; g.rrb = rrbias;
      gemm_wmma<EPI_QKV, 4><<<gemm_grid(g.M, g.N, 1, 4), 256, 0, stream>>>(g);
    }
    { // rk = R @ rW^T
      GemmArgs g = {};
      g.A = rpe; g.W = rwbb;
      g.M = KLEN; g.N = NHEAD * DHEAD; g.K = DMODEL; g.lda = DMODEL; g.ldw = DMODEL;
      g.outb = rkb;
      gemm_wmma<EPI_RK, 4><<<gemm_grid(g.M, g.N, 1, 4), 256, 0, stream>>>(g);
    }
    { // AC = (q+r_w_bias) @ K^T, batched over (b,n)
      GemmArgs g = {};
      g.A = qwb; g.W = kb;
      g.M = QLEN; g.N = KLEN; g.K = DHEAD; g.lda = DHEAD; g.ldw = DHEAD;
      g.sAz = (long long)QLEN * DHEAD; g.sWz = (long long)KLEN * DHEAD;
      g.outf = sc; g.ldc = KLEN; g.sCz = (long long)QLEN * KLEN;
      gemm_wmma<EPI_F32, 4><<<gemm_grid(g.M, g.N, BSZ * NHEAD, 4), 256, 0, stream>>>(g);
    }
    { // BD = (q+r_r_bias) @ rk^T, rk shared across batch (z % NHEAD)
      GemmArgs g = {};
      g.A = qrb; g.W = rkb;
      g.M = QLEN; g.N = KLEN; g.K = DHEAD; g.lda = DHEAD; g.ldw = DHEAD;
      g.sAz = (long long)QLEN * DHEAD; g.sWz = (long long)KLEN * DHEAD; g.wzmod = NHEAD;
      g.outf = bd; g.ldc = KLEN; g.sCz = (long long)QLEN * KLEN;
      gemm_wmma<EPI_F32, 4><<<gemm_grid(g.M, g.N, BSZ * NHEAD, 4), 256, 0, stream>>>(g);
    }
    softmax_kernel<<<dim3(QLEN, BSZ * NHEAD), 256, 0, stream>>>(sc, bd, (bf16*)sc);
    { // AV = P @ V   (P is bf16 aliased over sc with row stride 2*KLEN); N=64 -> NSUB=2
      GemmArgs g = {};
      g.A = (const bf16*)sc; g.W = vTb;
      g.M = QLEN; g.N = DHEAD; g.K = KLEN; g.lda = 2 * KLEN; g.ldw = KLEN;
      g.sAz = (long long)QLEN * 2 * KLEN; g.sWz = (long long)DHEAD * KLEN;
      g.outb = avb;
      gemm_wmma<EPI_AV, 2><<<gemm_grid(g.M, g.N, BSZ * NHEAD, 2), 256, 0, stream>>>(g);
    }
    { // output projection + residual
      GemmArgs g = {};
      g.A = avb; g.W = owb;
      g.M = NROWS; g.N = DMODEL; g.K = NHEAD * DHEAD;
      g.lda = NHEAD * DHEAD; g.ldw = NHEAD * DHEAD;
      g.outf = tmp; g.ldc = DMODEL; g.resid = h;
      gemm_wmma<EPI_RESID, 4><<<gemm_grid(g.M, g.N, 1, 4), 256, 0, stream>>>(g);
    }
    ln_kernel<<<NROWS, 256, 0, stream>>>(tmp, ln1g + (size_t)l * DMODEL,
                                         ln1b + (size_t)l * DMODEL, h, hb);
    { // FFN1: relu(h @ W1^T + b1) -> bf16
      GemmArgs g = {};
      g.A = hb; g.W = w1b;
      g.M = NROWS; g.N = DINNER; g.K = DMODEL; g.lda = DMODEL; g.ldw = DMODEL;
      g.outb = ff1; g.ldc = DINNER; g.bias = b1 + (size_t)l * DINNER;
      gemm_wmma<EPI_RELU_BF16, 4><<<gemm_grid(g.M, g.N, 1, 4), 256, 0, stream>>>(g);
    }
    { // FFN2: h + (ff1 @ W2^T + b2)
      GemmArgs g = {};
      g.A = ff1; g.W = w2b;
      g.M = NROWS; g.N = DMODEL; g.K = DINNER; g.lda = DINNER; g.ldw = DINNER;
      g.outf = tmp; g.ldc = DMODEL; g.resid = h; g.bias = b2 + (size_t)l * DMODEL;
      gemm_wmma<EPI_RESID_BIAS, 4><<<gemm_grid(g.M, g.N, 1, 4), 256, 0, stream>>>(g);
    }
    ln_kernel<<<NROWS, 256, 0, stream>>>(tmp, ln2g + (size_t)l * DMODEL,
                                         ln2b + (size_t)l * DMODEL, h, hb);
  }

  hipMemcpyAsync(d_out, h, (size_t)NROWS * DMODEL * 4, hipMemcpyDeviceToDevice, stream);
}
